// Qwen3Attention_88630945120688
// MI455X (gfx1250) — matrix operand order repaired and checked by reading, not compiled, not run on hardware
//
#include <hip/hip_runtime.h>

// ---------------- problem constants (Qwen3-ish attention block) ----------------
#define BB    2
#define TSEQ  2048
#define HID   4096
#define NH    32
#define NKV   8
#define HD    128
#define QKVN  ((NH + 2 * NKV) * HD)   // 6144
#define NTOK  (BB * TSEQ)             // 4096

typedef __attribute__((ext_vector_type(16))) __bf16 v16bf;
typedef __attribute__((ext_vector_type(8)))  float  v8f;
typedef __attribute__((ext_vector_type(4)))  int    v4i;

// ---- gfx1250 async global->LDS path (guarded; falls back to VGPR staging) ----
#if defined(__has_builtin)
#if __has_builtin(__builtin_amdgcn_global_load_async_to_lds_b128)
#define HAS_ASYNC_LDS 1
#endif
#endif
#ifndef HAS_ASYNC_LDS
#define HAS_ASYNC_LDS 0
#endif

#if HAS_ASYNC_LDS
typedef __attribute__((address_space(1))) v4i* as1_v4i;
typedef __attribute__((address_space(3))) v4i* as3_v4i;
__device__ __forceinline__ void async_copy_b128(const void* g, void* l) {
    __builtin_amdgcn_global_load_async_to_lds_b128((as1_v4i)(void*)g, (as3_v4i)l, 0, 0);
}
#if __has_builtin(__builtin_amdgcn_s_wait_asynccnt)
#define WAIT_ASYNC(n) __builtin_amdgcn_s_wait_asynccnt(n)
#else
#define WAIT_ASYNC(n) asm volatile("s_wait_asynccnt " #n ::: "memory")
#endif
#endif

union Frag {
    v16bf v;
    unsigned int u[8];
    uint4 q[2];
    unsigned short s[16];
};

__device__ __forceinline__ unsigned short f2bf(float f) {
    unsigned int u = __float_as_uint(f);
    return (unsigned short)((u + 0x7FFFu + ((u >> 16) & 1u)) >> 16);  // RNE
}

__device__ __forceinline__ v8f wmma_bf16(const Frag& a, const Frag& b, v8f c) {
    return __builtin_amdgcn_wmma_f32_16x16x32_bf16(false, a.v, false, b.v,
                                                   (short)0, c, false, false);
}

// A/P-style fragment (16xK A-matrix packing): two 16B chunks at +0 and +32 bytes
__device__ __forceinline__ void ld_frag_a16(Frag& f, const unsigned short* p) {
    f.q[0] = *(const uint4*)p;
    f.q[1] = *(const uint4*)(p + 16);
}
// B-style fragment (same packing as A): two 16B chunks at +0 and +32 bytes
__device__ __forceinline__ void ld_frag_b8(Frag& f, const unsigned short* p) {
    f.q[0] = *(const uint4*)p;
    f.q[1] = *(const uint4*)(p + 16);
}

// ---------------- fp32 -> bf16 pack (4 elems / thread) ----------------
__global__ __launch_bounds__(256)
void cvt_f32_bf16(const float4* __restrict__ in, uint2* __restrict__ out, int n4) {
    int i = blockIdx.x * blockDim.x + threadIdx.x;
    if (i < n4) {
        float4 v = in[i];
        uint2 o;
        o.x = (unsigned int)f2bf(v.x) | ((unsigned int)f2bf(v.y) << 16);
        o.y = (unsigned int)f2bf(v.z) | ((unsigned int)f2bf(v.w) << 16);
        out[i] = o;
    }
}

// ---------------- generic bf16 WMMA GEMM: C[M,N] = A[M,K] * B[K,N] ----------------
// A,B row-major bf16 (as u16); C fp32. 128 threads (4 waves); block tile 64x128,
// each wave owns 32x64 (2x4 of 16x16). Double-buffered LDS (2x24KB); A tile via
// async global->LDS b128 (ASYNCcnt) when available, B transposed in registers.
__global__ __launch_bounds__(128)
void gemm_bf16_wmma(const unsigned short* __restrict__ A,
                    const unsigned short* __restrict__ B,
                    float* __restrict__ C, int M, int N, int K) {
    __shared__ __align__(16) unsigned short As[2][64 * 64];    // [row][k]   2x 8 KB
    __shared__ __align__(16) unsigned short Bt[2][128 * 64];   // [n][k]     2x16 KB

    const int tid  = threadIdx.x;
    const int lane = tid & 31, wave = tid >> 5;
    const int lm = lane & 15, g = lane >> 4;
    const int wm = (wave >> 1) * 32;   // 0 / 32
    const int wn = (wave & 1) * 64;    // 0 / 64
    const int m0 = blockIdx.y * 64, n0 = blockIdx.x * 128;
    (void)M;

    const v8f vzero = {0.f, 0.f, 0.f, 0.f, 0.f, 0.f, 0.f, 0.f};
    v8f acc[2][4];
    for (int i = 0; i < 2; ++i)
        for (int j = 0; j < 4; ++j) acc[i][j] = vzero;

    // --- staging helpers ---
    uint4 rb[2][4];                    // B prefetch registers (2 blocks of 4k x 8n)
    auto loadB = [&](int kb) {
#pragma unroll
        for (int i = 0; i < 2; ++i) {
            int blk = i * 128 + tid;   // 16 k4 x 16 nq
            int k4 = blk & 15, nq = blk >> 4;
            const unsigned short* bp = B + (size_t)(kb + k4 * 4) * N + n0 + nq * 8;
#pragma unroll
            for (int j = 0; j < 4; ++j) rb[i][j] = *(const uint4*)(bp + j * N);
        }
    };
    auto storeB = [&](int buf) {
#pragma unroll
        for (int i = 0; i < 2; ++i) {
            int blk = i * 128 + tid;
            int k4 = blk & 15, nq = blk >> 4;
            int k = k4 * 4, nn = nq * 8;
            const unsigned short* s0 = (const unsigned short*)&rb[i][0];
            const unsigned short* s1 = (const unsigned short*)&rb[i][1];
            const unsigned short* s2 = (const unsigned short*)&rb[i][2];
            const unsigned short* s3 = (const unsigned short*)&rb[i][3];
#pragma unroll
            for (int j = 0; j < 8; ++j) {
                uint2 w;
                w.x = (unsigned int)s0[j] | ((unsigned int)s1[j] << 16);
                w.y = (unsigned int)s2[j] | ((unsigned int)s3[j] << 16);
                *(uint2*)(&Bt[buf][(nn + j) * 64 + k]) = w;
            }
        }
    };
#if HAS_ASYNC_LDS
    auto stageA = [&](int buf, int kb) {
#pragma unroll
        for (int i = 0; i < 4; ++i) {
            int idx = i * 128 + tid;   // 512 uint4 slots, 8 per row
            int row = idx >> 3, qq = idx & 7;
            async_copy_b128((const void*)(A + (size_t)(m0 + row) * K + kb + qq * 8),
                            (void*)(&As[buf][row * 64 + qq * 8]));
        }
    };
#else
    uint4 ra[4];
    auto loadA = [&](int kb) {
#pragma unroll
        for (int i = 0; i < 4; ++i) {
            int idx = i * 128 + tid;
            int row = idx >> 3, qq = idx & 7;
            ra[i] = *(const uint4*)(A + (size_t)(m0 + row) * K + kb + qq * 8);
        }
    };
    auto storeA = [&](int buf) {
#pragma unroll
        for (int i = 0; i < 4; ++i) {
            int idx = i * 128 + tid;
            int row = idx >> 3, qq = idx & 7;
            ((uint4*)&As[buf][0])[row * 8 + qq] = ra[i];
        }
    };
#endif

    // --- prologue: fetch first tile ---
    loadB(0);
#if HAS_ASYNC_LDS
    stageA(0, 0);
#else
    loadA(0);
#endif

    int buf = 0;
    for (int kb = 0; kb < K; kb += 64) {
        storeB(buf);
#if !HAS_ASYNC_LDS
        storeA(buf);
#endif
        const bool more = (kb + 64) < K;
        if (more) {
#if HAS_ASYNC_LDS
            stageA(buf ^ 1, kb + 64);   // next tile's async copies in flight
#else
            loadA(kb + 64);
#endif
            loadB(kb + 64);
        }
#if HAS_ASYNC_LDS
        if (more) { WAIT_ASYNC(4); }    // current tile's 4 copies done; next 4 pending
        else      { WAIT_ASYNC(0); }
#endif
        __syncthreads();                // waits DScnt, then barrier

        const unsigned short* as = &As[buf][0];
        const unsigned short* bt = &Bt[buf][0];
#pragma unroll
        for (int kc = 0; kc < 64; kc += 32) {
            Frag af[2], bf[4];
#pragma unroll
            for (int t = 0; t < 2; ++t)
                ld_frag_a16(af[t], as + (wm + t * 16 + lm) * 64 + kc + g * 8);
#pragma unroll
            for (int t = 0; t < 4; ++t)
                ld_frag_b8(bf[t], bt + (wn + t * 16 + lm) * 64 + kc + g * 8);
#pragma unroll
            for (int i = 0; i < 2; ++i)
#pragma unroll
                for (int j = 0; j < 4; ++j)
                    acc[i][j] = wmma_bf16(af[i], bf[j], acc[i][j]);
        }
        __syncthreads();                // protect buf before next overwrite cycle
        buf ^= 1;
    }

    for (int i = 0; i < 2; ++i)
        for (int j = 0; j < 4; ++j)
#pragma unroll
            for (int r = 0; r < 8; ++r) {
                int row = m0 + wm + i * 16 + r + 8 * g;
                int col = n0 + wn + j * 16 + lm;
                C[(size_t)row * N + col] = acc[i][j][r];
            }
}

// ---------------- fused per-head RMSNorm + RoPE, fp32 -> bf16 ----------------
// 256 threads = 8 waves, one head per wave; grid (NTOK, 6) covers 48 heads.
// Q -> [B][H][T][D], K -> [B][KV][T][D], V -> transposed [B][KV][D][T].
__global__ __launch_bounds__(256)
void norm_rope(const float* __restrict__ qkv, const float* __restrict__ qs,
               const float* __restrict__ ks, const int* __restrict__ posn,
               unsigned short* __restrict__ Qb, unsigned short* __restrict__ Kb,
               unsigned short* __restrict__ Vt) {
    const int tok  = blockIdx.x;
    const int b    = tok / TSEQ, t = tok % TSEQ;
    const int wave = threadIdx.x >> 5, lane = threadIdx.x & 31;
    const int head = blockIdx.y * 8 + wave;
    const float* src = qkv + (size_t)tok * QKVN;
    const float  p   = (float)posn[tok];

    if (head < NH + NKV) {
        const float* xin;
        const float* sc;
        unsigned short* out;
        if (head < NH) {
            xin = src + head * HD;
            sc  = qs;
            out = Qb + (((size_t)(b * NH + head)) * TSEQ + t) * HD;
        } else {
            int kvh = head - NH;
            xin = src + HID + kvh * HD;
            sc  = ks;
            out = Kb + (((size_t)(b * NKV + kvh)) * TSEQ + t) * HD;
        }
        float x0 = xin[lane], x1 = xin[lane + 32], x2 = xin[lane + 64], x3 = xin[lane + 96];
        float ss = x0 * x0 + x1 * x1 + x2 * x2 + x3 * x3;
        for (int m = 1; m < 32; m <<= 1) ss += __shfl_xor(ss, m, 32);
        float r = rsqrtf(ss * (1.0f / HD) + 1e-6f);
        x0 *= r * sc[lane];      x1 *= r * sc[lane + 32];
        x2 *= r * sc[lane + 64]; x3 *= r * sc[lane + 96];
        // RoPE pairs (i, i+64): lane -> (x0,x2), lane+32 -> (x1,x3)
        float f0 = powf(1000000.0f, -((float)(2 * lane)) / 128.0f);
        float f1 = powf(1000000.0f, -((float)(2 * (lane + 32))) / 128.0f);
        float a0 = p * f0, a1 = p * f1;
        float c0 = cosf(a0), s0 = sinf(a0), c1 = cosf(a1), s1 = sinf(a1);
        out[lane]      = f2bf(x0 * c0 - x2 * s0);
        out[lane + 64] = f2bf(x2 * c0 + x0 * s0);
        out[lane + 32] = f2bf(x1 * c1 - x3 * s1);
        out[lane + 96] = f2bf(x3 * c1 + x1 * s1);
    } else {
        int kvh = head - NH - NKV;
        const float* xin = src + HID + NKV * HD + kvh * HD;
        unsigned short* out = Vt + ((size_t)(b * NKV + kvh)) * HD * TSEQ;
        out[(size_t)(lane)      * TSEQ + t] = f2bf(xin[lane]);
        out[(size_t)(lane + 32) * TSEQ + t] = f2bf(xin[lane + 32]);
        out[(size_t)(lane + 64) * TSEQ + t] = f2bf(xin[lane + 64]);
        out[(size_t)(lane + 96) * TSEQ + t] = f2bf(xin[lane + 96]);
    }
}

// ---------------- flash attention (causal, GQA) with WMMA ----------------
// 128 threads = 4 waves; each wave owns 16 query rows; block = 64-row q tile.
// Online softmax over 32-key steps; S and PV both via v_wmma_f32_16x16x32_bf16.
__global__ __launch_bounds__(128)
void flash_attn_wmma(const unsigned short* __restrict__ Qb,
                     const unsigned short* __restrict__ Kb,
                     const unsigned short* __restrict__ Vt,
                     unsigned short* __restrict__ Ob) {
    __shared__ __align__(16) unsigned short pbuf[4 * 16 * 32];  // per-wave P bounce

    const int lane = threadIdx.x & 31, wave = threadIdx.x >> 5;
    const int lm = lane & 15, g = lane >> 4;
    const int bh = blockIdx.x;
    const int b = bh / NH, h = bh % NH, kvh = h / (NH / NKV);
    const int q_lo = blockIdx.y * 64 + wave * 16;

    const unsigned short* Qrow  = Qb + (((size_t)(b * NH + h)) * TSEQ + q_lo) * HD;
    const unsigned short* Kbase = Kb + ((size_t)(b * NKV + kvh)) * TSEQ * HD;
    const unsigned short* Vbase = Vt + ((size_t)(b * NKV + kvh)) * HD * TSEQ;
    unsigned short* pb = pbuf + wave * 16 * 32;

    // Q A-fragments for this wave's 16 rows, 4 chunks of K=32 (D = 128)
    Frag qa[4];
    {
        const unsigned short* qr = Qrow + (size_t)lm * HD;
#pragma unroll
        for (int c = 0; c < 4; ++c) ld_frag_a16(qa[c], qr + c * 32 + g * 8);
    }

    const v8f vzero = {0.f, 0.f, 0.f, 0.f, 0.f, 0.f, 0.f, 0.f};
    v8f o[8];
    for (int c = 0; c < 8; ++c) o[c] = vzero;
    float m_[8], l_[8];
    for (int r = 0; r < 8; ++r) { m_[r] = -1e38f; l_[r] = 0.f; }
    const float scale = 0.08838834764831844f;  // 1/sqrt(128)

    const int kt_end = q_lo + 16;  // causal: keys needed up to q_lo+15
    for (int kt0 = 0; kt0 < kt_end; kt0 += 32) {
        // ---- S = Q * K^T for 32 keys (two 16x16 tiles) ----
        v8f s0 = vzero, s1 = vzero;
        const unsigned short* kr0 = Kbase + (size_t)(kt0 + lm) * HD + g * 8;
        const unsigned short* kr1 = Kbase + (size_t)(kt0 + 16 + lm) * HD + g * 8;
#pragma unroll
        for (int c = 0; c < 4; ++c) {
            Frag kb0, kb1;
            ld_frag_b8(kb0, kr0 + c * 32);
            ld_frag_b8(kb1, kr1 + c * 32);
            s0 = wmma_bf16(qa[c], kb0, s0);
            s1 = wmma_bf16(qa[c], kb1, s1);
        }

        // ---- causal mask + online softmax update ----
        float alpha[8];
#pragma unroll
        for (int r = 0; r < 8; ++r) {
            int row = q_lo + r + 8 * g;
            float v0 = (kt0 + lm      <= row) ? s0[r] * scale : -1e30f;
            float v1 = (kt0 + 16 + lm <= row) ? s1[r] * scale : -1e30f;
            float mr = fmaxf(v0, v1);
            for (int mm = 1; mm < 16; mm <<= 1) mr = fmaxf(mr, __shfl_xor(mr, mm, 32));
            float mnew = fmaxf(m_[r], mr);
            float a  = __expf(m_[r] - mnew);
            float e0 = __expf(v0 - mnew), e1 = __expf(v1 - mnew);
            float rs = e0 + e1;
            for (int mm = 1; mm < 16; mm <<= 1) rs += __shfl_xor(rs, mm, 32);
            l_[r] = l_[r] * a + rs;
            m_[r] = mnew;
            alpha[r] = a;
            s0[r] = e0;
            s1[r] = e1;
        }
#pragma unroll
        for (int c = 0; c < 8; ++c)
#pragma unroll
            for (int r = 0; r < 8; ++r) o[c][r] *= alpha[r];

        // ---- bounce P through LDS: C-layout -> A-layout (16x32 bf16) ----
#pragma unroll
        for (int r = 0; r < 8; ++r) {
            int row = r + 8 * g;
            pb[row * 32 + lm]      = f2bf(s0[r]);
            pb[row * 32 + 16 + lm] = f2bf(s1[r]);
        }
        asm volatile("s_wait_dscnt 0" ::: "memory");
        Frag pf;
        ld_frag_a16(pf, pb + lm * 32 + g * 8);

        // ---- O += P * V  (V stored [d][t]: contiguous keys) ----
#pragma unroll
        for (int c = 0; c < 8; ++c) {
            Frag vf;
            ld_frag_b8(vf, Vbase + (size_t)(c * 16 + lm) * TSEQ + kt0 + g * 8);
            o[c] = wmma_bf16(pf, vf, o[c]);
        }
    }

    // ---- epilogue: normalize and store bf16 rows [tok][h*D + d] ----
    for (int c = 0; c < 8; ++c)
#pragma unroll
        for (int r = 0; r < 8; ++r) {
            int row = q_lo + r + 8 * g;
            size_t tok = (size_t)b * TSEQ + row;
            Ob[tok * (NH * HD) + h * HD + c * 16 + lm] = f2bf(o[c][r] / l_[r]);
        }
}

// ---------------- host-side orchestration ----------------
extern "C" void kernel_launch(void* const* d_in, const int* in_sizes, int n_in,
                              void* d_out, int out_size, void* d_ws, size_t ws_size,
                              hipStream_t stream) {
    (void)in_sizes; (void)n_in; (void)out_size; (void)ws_size;
    const float* x       = (const float*)d_in[0];
    const float* w_qkv   = (const float*)d_in[1];
    const float* w_o     = (const float*)d_in[2];
    const float* q_scale = (const float*)d_in[3];
    const float* k_scale = (const float*)d_in[4];
    const int*   posn    = (const int*)d_in[5];
    // d_in[6] = attention_mask: all-true in this problem; causal mask dominates.

    char* ws = (char*)d_ws;
    size_t off = 0;
    auto alloc = [&](size_t bytes) { size_t o = off; off += (bytes + 255) & ~(size_t)255; return o; };
    const size_t o_xb    = alloc((size_t)NTOK * HID * 2);         // x bf16
    const size_t o_wqkvb = alloc((size_t)HID * QKVN * 2);         // w_qkv bf16
    const size_t o_wob   = alloc((size_t)HID * HID * 2);          // w_o bf16
    const size_t o_qkvf  = alloc((size_t)NTOK * QKVN * 4);        // qkv fp32
    const size_t o_Qb    = alloc((size_t)BB * NH * TSEQ * HD * 2);
    const size_t o_Kb    = alloc((size_t)BB * NKV * TSEQ * HD * 2);
    const size_t o_Vt    = alloc((size_t)BB * NKV * HD * TSEQ * 2);
    const size_t o_attn  = o_xb;  // alias: x-bf16 is dead once qkv GEMM is done

    unsigned short* xb    = (unsigned short*)(ws + o_xb);
    unsigned short* wqkvb = (unsigned short*)(ws + o_wqkvb);
    unsigned short* wob   = (unsigned short*)(ws + o_wob);
    float*          qkvf  = (float*)(ws + o_qkvf);
    unsigned short* Qb    = (unsigned short*)(ws + o_Qb);
    unsigned short* Kb    = (unsigned short*)(ws + o_Kb);
    unsigned short* Vt    = (unsigned short*)(ws + o_Vt);
    unsigned short* attnb = (unsigned short*)(ws + o_attn);

    // 1) fp32 -> bf16 conversions (b128 in / b64 out)
    {
        int n4 = NTOK * HID / 4;
        cvt_f32_bf16<<<(n4 + 255) / 256, 256, 0, stream>>>((const float4*)x, (uint2*)xb, n4);
        n4 = HID * QKVN / 4;
        cvt_f32_bf16<<<(n4 + 255) / 256, 256, 0, stream>>>((const float4*)w_qkv, (uint2*)wqkvb, n4);
        n4 = HID * HID / 4;
        cvt_f32_bf16<<<(n4 + 255) / 256, 256, 0, stream>>>((const float4*)w_o, (uint2*)wob, n4);
    }
    // 2) qkv = x @ w_qkv  [4096 x 6144]
    gemm_bf16_wmma<<<dim3(QKVN / 128, NTOK / 64), 128, 0, stream>>>(
        xb, wqkvb, qkvf, NTOK, QKVN, HID);
    // 3) per-head RMSNorm + RoPE, split into Q/K/V bf16 (V transposed)
    norm_rope<<<dim3(NTOK, 6), 256, 0, stream>>>(qkvf, q_scale, k_scale, posn, Qb, Kb, Vt);
    // 4) causal GQA flash attention
    flash_attn_wmma<<<dim3(BB * NH, TSEQ / 64), 128, 0, stream>>>(Qb, Kb, Vt, attnb);
    // 5) out = attn @ w_o  [4096 x 4096] fp32
    gemm_bf16_wmma<<<dim3(HID / 128, NTOK / 64), 128, 0, stream>>>(
        attnb, wob, (float*)d_out, NTOK, HID, NH * HD);
}
